// GSL_69320772157907
// MI455X (gfx1250) — matrix operand order repaired and checked by reading, not compiled, not run on hardware
//
#include <hip/hip_runtime.h>
#include <hip/hip_bf16.h>

// ---------------------------------------------------------------------------
// GSL forward for MI455X (gfx1250, wave32).
// sim = emb@embT via v_wmma_f32_16x16x32_f16 with 16x64 per-wave strips
// (register-level A reuse), top-21 fused (the NxN matrix is never stored).
// Normalized adjacency decomposed as e^-6 * u u^T + 420k-nnz sparse part.
// ---------------------------------------------------------------------------

#define NROWS 10000
#define DDIM  512
#define HDIM  64
#define KSEL  21
#define NTILI 625      // NROWS/16
#define NQUAD 157      // ceil(625/4)
#define EXPM6 0.00247875217666635842f

typedef _Float16 v8h  __attribute__((ext_vector_type(8)));
typedef _Float16 v16h __attribute__((ext_vector_type(16)));
typedef float    v8f  __attribute__((ext_vector_type(8)));

// A fragment, contiguous K: lane<16 gets K {0..15},
// lane>=16 gets K {16..31}; p already includes the per-lane koff (0/16).
__device__ __forceinline__ v16h load_a_frag(const _Float16* p) {
  v8h lo = *(const v8h*)(p);
  v8h hi = *(const v8h*)(p + 8);
  v16h r;
#pragma unroll
  for (int t = 0; t < 8; ++t) { r[t] = lo[t]; r[t + 8] = hi[t]; }
  return r;
}

// ---------------------------------------------------------------------------
// 1) h = tanh(x*dw1)*dw2, row L2-normalize -> emb (f16); also x -> f16.
__global__ __launch_bounds__(256) void k_prep_rows(
    const float* __restrict__ x, const float* __restrict__ dw1,
    const float* __restrict__ dw2, _Float16* __restrict__ embH,
    _Float16* __restrict__ xH) {
  int wave = threadIdx.x >> 5, lane = threadIdx.x & 31;
  int row = blockIdx.x * 8 + wave;
  const float* xr = x + (size_t)row * DDIM;
  float hv[16];
  float ss = 0.f;
#pragma unroll
  for (int t = 0; t < 16; ++t) {
    int d = t * 32 + lane;
    float h = tanhf(xr[d] * dw1[d]) * dw2[d];
    hv[t] = h;
    ss += h * h;
  }
#pragma unroll
  for (int m = 16; m >= 1; m >>= 1) ss += __shfl_xor(ss, m, 32);
  float rn = rsqrtf(ss);
#pragma unroll
  for (int t = 0; t < 16; ++t) {
    int d = t * 32 + lane;
    embH[(size_t)row * DDIM + d] = (_Float16)(hv[t] * rn);
    xH[(size_t)row * DDIM + d] = (_Float16)xr[d];
  }
}

// ---------------------------------------------------------------------------
// 2) Transpose W1,W2 to f16 row-per-output-column; zero colsum and g buffers.
__global__ __launch_bounds__(256) void k_prep_w_zero(
    const float* __restrict__ W1, const float* __restrict__ W2,
    _Float16* __restrict__ W1t, _Float16* __restrict__ W2t,
    float* __restrict__ colsum, float* __restrict__ gbuf) {
  int idx = blockIdx.x * 256 + threadIdx.x;
  if (idx < 32768) {                      // W1t[n][k] = W1[k][n], [64][512]
    int n = idx >> 9, k = idx & 511;
    W1t[idx] = (_Float16)W1[(size_t)k * 64 + n];
  } else if (idx < 36864) {               // W2t[n][k] = W2[k][n], [64][64]
    int j = idx - 32768;
    int n = j >> 6, k = j & 63;
    W2t[j] = (_Float16)W2[(size_t)k * 64 + n];
  } else if (idx < 46864) {
    colsum[idx - 36864] = 0.f;
  } else if (idx < 46992) {
    gbuf[idx - 46864] = 0.f;              // g1 (64) + g2 (64)
  }
}

// ---------------------------------------------------------------------------
// 3) sim = emb @ embT with fused per-row top-21 (value+index).
//    Block = 16 rows, 4 waves. Each wave processes a quad of 4 column tiles
//    (16x64 strip): one A fragment feeds 4 WMMAs per K-step. Epilogue dumps
//    the 4 tiles to LDS; lanes 0..15 insert 64 candidates into a sorted
//    per-row 21-list. Quad validity is wave-uniform (no EXEC divergence at
//    the WMMA site).
__global__ __launch_bounds__(128) void k_sim_topk(
    const _Float16* __restrict__ embH, float* __restrict__ topv,
    int* __restrict__ topi) {
  __shared__ float tile[4][4][256];       // [wave][q][16x16]
  __shared__ float listv[4][16][KSEL];
  __shared__ int   listj[4][16][KSEL];

  int tid = threadIdx.x, wave = tid >> 5, lane = tid & 31;
  for (int t = tid; t < 4 * 16 * KSEL; t += 128) {
    ((float*)listv)[t] = -3.0e38f;
    ((int*)listj)[t] = 0x7fffffff;
  }
  __syncthreads();

  int i0 = blockIdx.x * 16;
  int lr = lane & 15;
  int koffA = (lane >> 4) * 16;           // A: contiguous 16 halves
  int koffB = (lane >> 4) * 16;           // B: contiguous 16 halves
  const _Float16* Ap = embH + (size_t)(i0 + lr) * DDIM + koffA;

  for (int jq = wave; jq < NQUAD; jq += 4) {
    int j0 = jq * 64;
    const _Float16* Bp[4];
#pragma unroll
    for (int q = 0; q < 4; ++q) {
      int jr = j0 + q * 16 + lr;
      if (jr >= NROWS) jr = NROWS - 1;    // clamp (remainder quad)
      Bp[q] = embH + (size_t)jr * DDIM + koffB;
    }
    v8f acc[4];
#pragma unroll
    for (int q = 0; q < 4; ++q) acc[q] = (v8f){};
#pragma unroll 2
    for (int k0 = 0; k0 < DDIM; k0 += 32) {
      v16h a = load_a_frag(Ap + k0);
#pragma unroll
      for (int q = 0; q < 4; ++q) {
        v16h b = *(const v16h*)(Bp[q] + k0);
        acc[q] = __builtin_amdgcn_wmma_f32_16x16x32_f16(
            false, a, false, b, (short)0, acc[q], false, false);
      }
    }
    int mb = (lane >> 4) * 8, nn = lane & 15;
#pragma unroll
    for (int q = 0; q < 4; ++q)
#pragma unroll
      for (int v = 0; v < 8; ++v) tile[wave][q][(mb + v) * 16 + nn] = acc[q][v];
    __builtin_amdgcn_wave_barrier();
    asm volatile("" ::: "memory");        // LDS per-wave in-order; fence compiler
    if (lane < 16) {
      float* Lv = listv[wave][lane];
      int*   Lj = listj[wave][lane];
      for (int q = 0; q < 4; ++q) {
        if (jq * 4 + q >= NTILI) break;   // wave-uniform guard
        for (int n = 0; n < 16; ++n) {
          float v = tile[wave][q][lane * 16 + n];
          int j = j0 + q * 16 + n;
          if (v > Lv[KSEL - 1]) {
            int p = KSEL - 1;
            while (p > 0 && Lv[p - 1] < v) {
              Lv[p] = Lv[p - 1]; Lj[p] = Lj[p - 1]; --p;
            }
            Lv[p] = v; Lj[p] = j;
          }
        }
      }
    }
    __builtin_amdgcn_wave_barrier();
    asm volatile("" ::: "memory");
  }
  __syncthreads();
  // Merge 4 x 21 candidates per row -> global top-21 (ties: lower index first).
  if (tid < 16) {
    int r = tid;
    for (int sel = 0; sel < KSEL; ++sel) {
      float best = -3.2e38f; int bj = 0x7fffffff, bw = 0, bp = 0;
      for (int w = 0; w < 4; ++w)
        for (int p = 0; p < KSEL; ++p) {
          float v = listv[w][r][p]; int j = listj[w][r][p];
          if (v > best || (v == best && j < bj)) { best = v; bj = j; bw = w; bp = p; }
        }
      topv[(size_t)(i0 + r) * KSEL + sel] = best;
      topi[(size_t)(i0 + r) * KSEL + sel] = bj;
      listv[bw][r][bp] = -3.3e38f;
      listj[bw][r][bp] = 0x7fffffff;
    }
  }
}

// ---------------------------------------------------------------------------
// 4) Per-kept-entry S value = (elu(6s-6)+1) - e^-6; rowsum; atomic colsum.
__global__ __launch_bounds__(256) void k_degree(
    const float* __restrict__ topv, const int* __restrict__ topi,
    float* __restrict__ sval, float* __restrict__ rowsum,
    float* __restrict__ colsum) {
  int i = blockIdx.x * 256 + threadIdx.x;
  if (i >= NROWS) return;
  float rs = 0.f;
  for (int t = 0; t < KSEL; ++t) {
    float s = topv[(size_t)i * KSEL + t];
    int j = topi[(size_t)i * KSEL + t];
    float z = 6.f * s - 6.f;
    float a = (z > 0.f) ? (z + 1.f) : __expf(z);
    float sv = a - EXPM6;
    sval[(size_t)i * KSEL + t] = sv;
    rs += sv;
    atomicAdd(colsum + j, sv);
  }
  rowsum[i] = rs;
}

// 5) u = d^{-1/2},  d_i = N*e^-6 + 0.5*(rowsum_i + colsum_i)
__global__ __launch_bounds__(256) void k_finish_u(
    const float* __restrict__ rowsum, const float* __restrict__ colsum,
    float* __restrict__ u) {
  int i = blockIdx.x * 256 + threadIdx.x;
  if (i >= NROWS) return;
  u[i] = rsqrtf((float)NROWS * EXPM6 + 0.5f * (rowsum[i] + colsum[i]));
}

// ---------------------------------------------------------------------------
// 6) WMMA GEMM: Out[N][64] = A[N][K] @ Bt[64][K]^T + bias.
//    One wave per 16-row strip computes all 64 columns (4 accumulators),
//    so each A fragment is loaded once per K-step and feeds 4 WMMAs.
__global__ __launch_bounds__(128) void k_gemm_f16(
    const _Float16* __restrict__ A, const _Float16* __restrict__ Bt,
    const float* __restrict__ bias, float* __restrict__ Out, int K) {
  int wave = threadIdx.x >> 5, lane = threadIdx.x & 31;
  int ti = blockIdx.x * 4 + wave;         // 16-row tile index (wave-uniform)
  if (ti >= NTILI) return;
  int i0 = ti * 16;
  int lr = lane & 15;
  const _Float16* Ap = A + (size_t)(i0 + lr) * K + ((lane >> 4) * 16);
  const _Float16* Bp[4];
#pragma unroll
  for (int q = 0; q < 4; ++q)
    Bp[q] = Bt + (size_t)(q * 16 + lr) * K + ((lane >> 4) * 16);
  v8f acc[4];
#pragma unroll
  for (int q = 0; q < 4; ++q) acc[q] = (v8f){};
  for (int k0 = 0; k0 < K; k0 += 32) {
    v16h a = load_a_frag(Ap + k0);
#pragma unroll
    for (int q = 0; q < 4; ++q) {
      v16h b = *(const v16h*)(Bp[q] + k0);
      acc[q] = __builtin_amdgcn_wmma_f32_16x16x32_f16(
          false, a, false, b, (short)0, acc[q], false, false);
    }
  }
  int mb = (lane >> 4) * 8, nn = lane & 15;
#pragma unroll
  for (int q = 0; q < 4; ++q) {
    int col = q * 16 + nn;
    float bc = bias[col];
#pragma unroll
    for (int v = 0; v < 8; ++v)
      Out[(size_t)(i0 + mb + v) * HDIM + col] = acc[q][v] + bc;
  }
}

// 7) g[o] = sum_i u_i * Y[i][o]
__global__ __launch_bounds__(256) void k_rank1(
    const float* __restrict__ Y, const float* __restrict__ u,
    float* __restrict__ g) {
  int tid = threadIdx.x;
  int col = tid & 63;
  int r0 = blockIdx.x * 4 + (tid >> 6);
  float s = 0.f;
  for (int r = r0; r < NROWS; r += 256) s += u[r] * Y[(size_t)r * HDIM + col];
  atomicAdd(g + col, s);
}

// 8) Z[i][o] = e^-6 * u_i * g[o]   (rank-1 part of adjN @ Y)
__global__ __launch_bounds__(256) void k_zinit(
    float* __restrict__ Z, const float* __restrict__ u,
    const float* __restrict__ g) {
  int idx = blockIdx.x * 256 + threadIdx.x;
  if (idx >= NROWS * HDIM) return;
  int i = idx >> 6, o = idx & 63;
  Z[idx] = EXPM6 * u[i] * g[o];
}

// 9) Sparse symmetric scatter: for each S nonzero (i,j,v):
//    coef = 0.5*u_i*u_j*v; Out[i]+=coef*Y[j]; Out[j]+=coef*Y[i].
//    Diagonal (i==j) double-counts -> exactly u^2 * S_ii, which is correct.
__global__ __launch_bounds__(256) void k_scatter(
    const int* __restrict__ topi, const float* __restrict__ sval,
    const float* __restrict__ u, const float* __restrict__ Y,
    float* __restrict__ Out) {
  int e = blockIdx.x * 8 + (threadIdx.x >> 5);
  int lane = threadIdx.x & 31;
  int i = e / KSEL;
  int j = topi[e];
  float c = 0.5f * u[i] * u[j] * sval[e];
  const float* Yi = Y + (size_t)i * HDIM;
  const float* Yj = Y + (size_t)j * HDIM;
  float* Oi = Out + (size_t)i * HDIM;
  float* Oj = Out + (size_t)j * HDIM;
  atomicAdd(Oi + lane, c * Yj[lane]);
  atomicAdd(Oi + lane + 32, c * Yj[lane + 32]);
  atomicAdd(Oj + lane, c * Yi[lane]);
  atomicAdd(Oj + lane + 32, c * Yi[lane + 32]);
}

// 10) h1 = relu(Z1) as f16 for the second WMMA GEMM.
__global__ __launch_bounds__(256) void k_relu_cvt(
    const float* __restrict__ Z, _Float16* __restrict__ Hh) {
  int idx = blockIdx.x * 256 + threadIdx.x;
  if (idx >= NROWS * HDIM) return;
  Hh[idx] = (_Float16)fmaxf(Z[idx], 0.f);
}

// ---------------------------------------------------------------------------
extern "C" void kernel_launch(void* const* d_in, const int* in_sizes, int n_in,
                              void* d_out, int out_size, void* d_ws, size_t ws_size,
                              hipStream_t stream) {
  (void)in_sizes; (void)n_in; (void)out_size; (void)ws_size;
  const float* x   = (const float*)d_in[0];
  const float* dw1 = (const float*)d_in[1];
  const float* dw2 = (const float*)d_in[2];
  const float* W1  = (const float*)d_in[3];
  const float* b1  = (const float*)d_in[4];
  const float* W2  = (const float*)d_in[5];
  const float* b2  = (const float*)d_in[6];
  float* out = (float*)d_out;

  char* p = (char*)d_ws;
  auto alloc = [&](size_t bytes) {
    char* r = p;
    p += (bytes + 1023) & ~(size_t)1023;
    return r;
  };
  _Float16* embH = (_Float16*)alloc((size_t)NROWS * DDIM * 2);   // 10.24 MB
  _Float16* xH   = (_Float16*)alloc((size_t)NROWS * DDIM * 2);   // 10.24 MB
  _Float16* W1t  = (_Float16*)alloc((size_t)HDIM * DDIM * 2);
  _Float16* W2t  = (_Float16*)alloc((size_t)HDIM * HDIM * 2);
  float* topv    = (float*)alloc((size_t)NROWS * KSEL * 4);
  int*   topi    = (int*)alloc((size_t)NROWS * KSEL * 4);
  float* sval    = (float*)alloc((size_t)NROWS * KSEL * 4);
  float* rowsum  = (float*)alloc((size_t)NROWS * 4);
  float* colsum  = (float*)alloc((size_t)NROWS * 4);
  float* uvec    = (float*)alloc((size_t)NROWS * 4);
  float* gbuf    = (float*)alloc(128 * 4);                       // g1|g2
  float* g1 = gbuf, *g2 = gbuf + 64;
  float* Y1      = (float*)alloc((size_t)NROWS * HDIM * 4);
  float* Z1      = (float*)alloc((size_t)NROWS * HDIM * 4);
  _Float16* h1H  = (_Float16*)alloc((size_t)NROWS * HDIM * 2);
  float* Y2      = (float*)alloc((size_t)NROWS * HDIM * 4);

  // 1. preprocess rows -> emb f16, x f16
  k_prep_rows<<<NROWS / 8, 256, 0, stream>>>(x, dw1, dw2, embH, xH);
  // 2. weights transpose + zero colsum/g
  k_prep_w_zero<<<184, 256, 0, stream>>>(W1, W2, W1t, W2t, colsum, gbuf);
  // 3. sim + fused top-21 (WMMA, 16x64 strips)
  k_sim_topk<<<NTILI, 128, 0, stream>>>(embH, topv, topi);
  // 4. S values, rowsum, colsum
  k_degree<<<(NROWS + 255) / 256, 256, 0, stream>>>(topv, topi, sval, rowsum, colsum);
  // 5. u = d^{-1/2}
  k_finish_u<<<(NROWS + 255) / 256, 256, 0, stream>>>(rowsum, colsum, uvec);
  // 6. Y1 = x @ W1 + b1 (WMMA)
  k_gemm_f16<<<NQUAD, 128, 0, stream>>>(xH, W1t, b1, Y1, DDIM);
  // 7-9. Z1 = adjN @ Y1 = e^-6 u (u^T Y1) + P @ Y1
  k_rank1<<<64, 256, 0, stream>>>(Y1, uvec, g1);
  k_zinit<<<(NROWS * HDIM) / 256, 256, 0, stream>>>(Z1, uvec, g1);
  k_scatter<<<(NROWS * KSEL) / 8, 256, 0, stream>>>(topi, sval, uvec, Y1, Z1);
  // 10. h1 = relu(Z1) -> f16
  k_relu_cvt<<<(NROWS * HDIM) / 256, 256, 0, stream>>>(Z1, h1H);
  // 11. Y2 = h1 @ W2 + b2 (WMMA)
  k_gemm_f16<<<NQUAD, 128, 0, stream>>>(h1H, W2t, b2, Y2, HDIM);
  // 12-14. out = adjN @ Y2
  k_rank1<<<64, 256, 0, stream>>>(Y2, uvec, g2);
  k_zinit<<<(NROWS * HDIM) / 256, 256, 0, stream>>>(out, uvec, g2);
  k_scatter<<<(NROWS * KSEL) / 8, 256, 0, stream>>>(topi, sval, uvec, Y2, out);
}
